// GlobalPointer_74895639707807
// MI455X (gfx1250) — matrix operand order repaired and checked by reading, not compiled, not run on hardware
//
#include <hip/hip_runtime.h>
#include <hip/hip_bf16.h>

// ---------------------------------------------------------------------------
// Problem constants (from the reference)
// ---------------------------------------------------------------------------
#define HEADS     12
#define HEAD_SIZE 64
#define HIDDEN    768
#define BB        4
#define SS        2048
#define OUTCH     (HEADS * 2 * HEAD_SIZE)   // 1536
#define MTOT      (BB * SS)                 // 8192 rows in projection GEMM

typedef __attribute__((ext_vector_type(16))) __bf16 v16bf;
typedef __attribute__((ext_vector_type(8)))  float  v8f;

// ---------------------------------------------------------------------------
// Kernel 1: QKV projection  proj[m,o] = sum_k X[m,k] * W[o,k] + b[o]
// M = 8192, N = 1536, K = 768.  One wave computes a 32x32 tile (2x2 WMMA
// 16x16 sub-tiles, f32 accumulate, bf16 inputs).  Results are de-interleaved
// into Q and K bf16 workspace tensors laid out (B, H, S, 64).
//
// bf16 A-fragment lane layout (16x32, MxK), wave32:
//   lane L (m = L&15, half = L>>4):
//     elems 0..15 -> K = half*16 + e
// bf16 B-fragment lane layout (32x16, KxN):
//   lane L (n = L&15, half = L>>4): elems 0..15 -> K = half*16 + e
// C/D layout: lane L (n = L&15, half = L>>4), VGPR v -> (m = half*8 + v, n)
// ---------------------------------------------------------------------------
__global__ __launch_bounds__(256) void qkv_proj_wmma_kernel(
    const float* __restrict__ X,      // (B*S, HIDDEN)
    const float* __restrict__ W,      // (OUTCH, HIDDEN)
    const float* __restrict__ bias,   // (OUTCH)
    __bf16* __restrict__ q_ws,        // (B, H, S, 64)
    __bf16* __restrict__ k_ws)        // (B, H, S, 64)
{
    const int lane = threadIdx.x & 31;
    const int wave = threadIdx.x >> 5;
    const int half = lane >> 4;
    const int l16  = lane & 15;

    const int NT   = OUTCH / 32;                 // 48 tile columns
    const int tile = blockIdx.x * 8 + wave;      // 12288 tiles total
    const int m0   = (tile / NT) * 32;
    const int n0   = (tile % NT) * 32;

    // Accumulators pre-loaded with bias (column is constant per lane).
    v8f acc[2][2];
#pragma unroll
    for (int mt = 0; mt < 2; ++mt)
#pragma unroll
        for (int nt = 0; nt < 2; ++nt) {
            const float bv = bias[n0 + nt * 16 + l16];
#pragma unroll
            for (int v = 0; v < 8; ++v) acc[mt][nt][v] = bv;
        }

    for (int kb = 0; kb < HIDDEN; kb += 32) {
        if (kb + 32 < HIDDEN) {  // prefetch next K-slab (global_prefetch_b8)
            __builtin_prefetch(X + (size_t)(m0 + l16) * HIDDEN + kb + 32, 0, 1);
            __builtin_prefetch(W + (size_t)(n0 + l16) * HIDDEN + kb + 32, 0, 1);
        }

        v16bf afrag[2], bfrag[2];
#pragma unroll
        for (int mt = 0; mt < 2; ++mt) {
            const float* ar = X + (size_t)(m0 + mt * 16 + l16) * HIDDEN + kb;
#pragma unroll
            for (int e = 0; e < 8; ++e) {
                afrag[mt][e]     = (__bf16)ar[half * 16 + e];
                afrag[mt][8 + e] = (__bf16)ar[half * 16 + 8 + e];
            }
        }
#pragma unroll
        for (int nt = 0; nt < 2; ++nt) {
            const float* br = W + (size_t)(n0 + nt * 16 + l16) * HIDDEN + kb + half * 16;
#pragma unroll
            for (int e = 0; e < 16; ++e) bfrag[nt][e] = (__bf16)br[e];
        }

#pragma unroll
        for (int mt = 0; mt < 2; ++mt)
#pragma unroll
            for (int nt = 0; nt < 2; ++nt)
                acc[mt][nt] = __builtin_amdgcn_wmma_f32_16x16x32_bf16(
                    false, afrag[mt], false, bfrag[nt],
                    (short)0, acc[mt][nt], false, false);
    }

    // De-interleave into Q / K bf16 workspace.  A 16-wide aligned column group
    // never crosses a 64-channel q/k block, so h/sel are uniform per sub-tile.
#pragma unroll
    for (int mt = 0; mt < 2; ++mt) {
#pragma unroll
        for (int nt = 0; nt < 2; ++nt) {
            const int o   = n0 + nt * 16 + l16;   // output channel
            const int h   = o >> 7;               // head
            const int sel = (o >> 6) & 1;         // 0 = Q, 1 = K
            const int d   = o & 63;               // head dim
            __bf16* dst = sel ? k_ws : q_ws;
#pragma unroll
            for (int v = 0; v < 8; ++v) {
                const int row = m0 + mt * 16 + half * 8 + v;  // [0, B*S)
                const int b   = row >> 11;
                const int s   = row & (SS - 1);
                const size_t idx =
                    (((size_t)b * HEADS + h) * SS + s) * HEAD_SIZE + d;
                dst[idx] = (__bf16)acc[mt][nt][v];
            }
        }
    }
}

// ---------------------------------------------------------------------------
// Kernel 2: logits[b,h,m,n] = (Q[b,h,m,:] . K[b,h,n,:]) / 8
//                             - 10000*(1 - a[m]*a[n]) - 10000*tril(m,n)
// Per (b,h): M = N = 2048, K = 64 (two bf16 WMMA K-steps).  One wave per
// 32x32 output tile.  This kernel is the bandwidth hog (805 MB out).
// ---------------------------------------------------------------------------
__global__ __launch_bounds__(256) void attn_logits_wmma_kernel(
    const __bf16* __restrict__ q_ws,
    const __bf16* __restrict__ k_ws,
    const int* __restrict__ attn,     // (B, S)
    float* __restrict__ out)          // (B, H, S, S)
{
    const int lane = threadIdx.x & 31;
    const int wave = threadIdx.x >> 5;
    const int half = lane >> 4;
    const int l16  = lane & 15;

    const int TPBH = (SS / 32) * (SS / 32);      // 4096 tiles per (b,h)
    const int tile = blockIdx.x * 8 + wave;      // 196608 tiles total
    const int bh   = tile / TPBH;
    const int t    = tile % TPBH;
    const int b    = bh / HEADS;
    const int m0   = (t / (SS / 32)) * 32;
    const int n0   = (t % (SS / 32)) * 32;

    const __bf16* qb = q_ws + (size_t)bh * SS * HEAD_SIZE;
    const __bf16* kb = k_ws + (size_t)bh * SS * HEAD_SIZE;

    v8f acc[2][2];
#pragma unroll
    for (int mt = 0; mt < 2; ++mt)
#pragma unroll
        for (int nt = 0; nt < 2; ++nt)
#pragma unroll
            for (int v = 0; v < 8; ++v) acc[mt][nt][v] = 0.0f;

#pragma unroll
    for (int ks = 0; ks < HEAD_SIZE; ks += 32) {
        v16bf afrag[2], bfrag[2];
#pragma unroll
        for (int mt = 0; mt < 2; ++mt) {
            const __bf16* ar =
                qb + (size_t)(m0 + mt * 16 + l16) * HEAD_SIZE + ks;
#pragma unroll
            for (int e = 0; e < 8; ++e) {
                afrag[mt][e]     = ar[half * 16 + e];
                afrag[mt][8 + e] = ar[half * 16 + 8 + e];
            }
        }
#pragma unroll
        for (int nt = 0; nt < 2; ++nt) {
            const __bf16* br =
                kb + (size_t)(n0 + nt * 16 + l16) * HEAD_SIZE + ks + half * 16;
#pragma unroll
            for (int e = 0; e < 16; ++e) bfrag[nt][e] = br[e];
        }

#pragma unroll
        for (int mt = 0; mt < 2; ++mt)
#pragma unroll
            for (int nt = 0; nt < 2; ++nt)
                acc[mt][nt] = __builtin_amdgcn_wmma_f32_16x16x32_bf16(
                    false, afrag[mt], false, bfrag[nt],
                    (short)0, acc[mt][nt], false, false);
    }

    // Epilogue: scale + pad mask + causal mask, stream to HBM.
    const int* am = attn + b * SS;
#pragma unroll
    for (int mt = 0; mt < 2; ++mt) {
#pragma unroll
        for (int nt = 0; nt < 2; ++nt) {
            const int   n   = n0 + nt * 16 + l16;
            const float anf = (float)am[n];
#pragma unroll
            for (int v = 0; v < 8; ++v) {
                const int   m   = m0 + mt * 16 + half * 8 + v;
                const float amf = (float)am[m];
                const float pm  = 1.0f - amf * anf;
                const float low = (n <= m) ? 1.0f : 0.0f;
                const float val =
                    acc[mt][nt][v] * 0.125f - 10000.0f * pm - 10000.0f * low;
                out[((size_t)bh * SS + m) * SS + n] = val;
            }
        }
    }
}

// ---------------------------------------------------------------------------
// Kernel 3: pair_mask_out[b,0,m,n] = (a[m]*a[n] == 0) || (n <= m)
// Written as 0.0/1.0 floats.  Pure streaming write, float4 per thread.
// ---------------------------------------------------------------------------
__global__ __launch_bounds__(256) void mask_out_kernel(
    const int* __restrict__ attn, float* __restrict__ out_mask)
{
    const size_t tid = (size_t)blockIdx.x * blockDim.x + threadIdx.x;
    const size_t e0  = tid * 4;                  // 4 n-values per thread
    const int    n0  = (int)(e0 & (SS - 1));
    const size_t row = e0 >> 11;                 // / SS
    const int    m   = (int)(row & (SS - 1));
    const int    b   = (int)(row >> 11);

    const int amv = attn[b * SS + m];
    float4 v;
    float* vp = (float*)&v;
#pragma unroll
    for (int j = 0; j < 4; ++j) {
        const int n  = n0 + j;
        const int an = attn[b * SS + n];
        vp[j] = ((amv * an == 0) || (n <= m)) ? 1.0f : 0.0f;
    }
    *(float4*)(out_mask + e0) = v;
}

// ---------------------------------------------------------------------------
// Host launcher
// ---------------------------------------------------------------------------
extern "C" void kernel_launch(void* const* d_in, const int* in_sizes, int n_in,
                              void* d_out, int out_size, void* d_ws, size_t ws_size,
                              hipStream_t stream) {
    const float* X    = (const float*)d_in[0];   // (B, S, HIDDEN)
    const int*   attn = (const int*)d_in[1];     // (B, S)
    const float* W    = (const float*)d_in[2];   // (OUTCH, HIDDEN)
    const float* bias = (const float*)d_in[3];   // (OUTCH)

    float* logits   = (float*)d_out;                               // (B,H,S,S)
    float* mask_out = logits + (size_t)BB * HEADS * SS * SS;       // (B,1,S,S)

    __bf16* q_ws = (__bf16*)d_ws;                                  // (B,H,S,64)
    __bf16* k_ws = q_ws + (size_t)BB * HEADS * SS * HEAD_SIZE;

    // K1: 8192/32 * 1536/32 = 12288 tiles, 8 waves/block -> 1536 blocks
    qkv_proj_wmma_kernel<<<(MTOT / 32) * (OUTCH / 32) / 8, 256, 0, stream>>>(
        X, W, bias, q_ws, k_ws);

    // K2: 48 * (64*64) = 196608 tiles, 8 waves/block -> 24576 blocks
    attn_logits_wmma_kernel<<<BB * HEADS * (SS / 32) * (SS / 32) / 8, 256, 0,
                              stream>>>(q_ws, k_ws, attn, logits);

    // K3: B*S*S/4 threads -> 16384 blocks of 256
    mask_out_kernel<<<(BB * SS * SS / 4) / 256, 256, 0, stream>>>(attn, mask_out);
}
